// GPN_3633542333124
// MI455X (gfx1250) — matrix operand order repaired and checked by reading, not compiled, not run on hardware
//
#include <hip/hip_runtime.h>
#include <math.h>
#include <stdint.h>

// ---------------- problem constants ----------------
#define DD 128
constexpr int Bb = 8;
constexpr int Nn = 5000;
constexpr int Ee = 40000;

typedef __attribute__((ext_vector_type(16))) _Float16 v16h;
typedef __attribute__((ext_vector_type(8)))  float    v8f;
typedef __attribute__((ext_vector_type(4)))  unsigned v4u;
typedef __attribute__((ext_vector_type(8)))  int      v8i;
typedef __attribute__((ext_vector_type(4)))  int      v4i;

union AF { v16h v; uint4 q[2]; };

// A fragment (16x32 f16, row-major source A[m][k], 128 cols):
// lanes 0-15: m=lane, VGPR0-3 = K k0..k0+7, VGPR4-7 = K k0+16..k0+23
// lanes 16-31: m=lane-16, VGPR0-3 = K k0+8..k0+15, VGPR4-7 = K k0+24..k0+31
__device__ inline v16h load_a_frag(const _Float16* A, int m0, int k0, int lane) {
    int m    = m0 + (lane & 15);
    int koff = (lane >> 4) * 8;
    AF f;
    const _Float16* p = A + (size_t)m * DD + k0 + koff;
    f.q[0] = *reinterpret_cast<const uint4*>(p);
    f.q[1] = *reinterpret_cast<const uint4*>(p + 16);
    return f.v;
}

// B fragment (32x16 f16) from transposed weight Wt[n][k] (128x128):
// lane holds column n = lane&15; lanes 0-15 K=k0..k0+7,k0+16..k0+23, lanes 16-31 K=k0+8..k0+15,k0+24..k0+31
__device__ inline v16h load_b_frag(const _Float16* Wt, int n0, int k0, int lane) {
    int n  = n0 + (lane & 15);
    int kb = k0 + (lane >> 4) * 8;
    AF f;
    const _Float16* p = Wt + (size_t)n * DD + kb;
    f.q[0] = *reinterpret_cast<const uint4*>(p);
    f.q[1] = *reinterpret_cast<const uint4*>(p + 16);
    return f.v;
}

// ---- TDM: DMA a 16x128 f16 strip (4 KB) from global into LDS at lds_off ----
// D# per CDNA5 ISA. 2D tensor == tile: dim0=128 elems (2B),
// dim1=16 rows, stride0=128. group0: count=1, type=2, global tile addr.
// This toolchain exposes the 6-arg builtin: (v4u, v8i, v4i, v4i, v8i, i32 cpol).
__device__ inline void tdm_load_tile16(const _Float16* gsrc, unsigned lds_off) {
    unsigned long long ga = (unsigned long long)(uintptr_t)gsrc;
    v4u g0;
    g0[0] = 1u;                                                 // count=1
    g0[1] = lds_off;                                            // lds_addr
    g0[2] = (unsigned)ga;                                       // global_addr[31:0]
    g0[3] = (unsigned)((ga >> 32) & 0x01FFFFFFull) | (2u << 30);// addr[56:32] | type=2
    v8i g1 = { (int)0x00010000u,        // data_size=1 (2 bytes), wg_mask=0
               (int)(128u << 16),       // tensor_dim0 = 128 (low16 at [63:48])
               (int)(16u  << 16),       // tensor_dim1 = 16 (low16 at [95:80])
               (int)(128u << 16),       // tile_dim0 = 128 at [127:112]
               16,                      // tile_dim1 = 16 at [143:128]
               128,                     // tensor_dim0_stride low32 = 128
               0, 0 };
    v4i z4 = {0, 0, 0, 0};
    v8i z8 = {0, 0, 0, 0, 0, 0, 0, 0};
    __builtin_amdgcn_tensor_load_to_lds(g0, g1, z4, z4, z8, 0);
}

// ---------------- 1. node embedding (K=2), zero neigh/cnt ----------------
__global__ void k_embed(const float* __restrict__ nc, const float* __restrict__ Wn,
                        const float* __restrict__ bn,
                        float* emb, _Float16* emb_h, float* neigh, float* cnt) {
    int i = blockIdx.x, d = threadIdx.x;
    float v = nc[(size_t)i * 2 + 0] * Wn[d] + nc[(size_t)i * 2 + 1] * Wn[DD + d] + bn[d];
    emb[(size_t)i * DD + d]   = v;
    emb_h[(size_t)i * DD + d] = (_Float16)v;
    neigh[(size_t)i * DD + d] = 0.f;
    if (d == 0) cnt[i] = 0.f;
}

// ---------------- 2. scatter-add neighbor aggregation ----------------
__global__ void k_scatter(const int* __restrict__ eidx, const float* __restrict__ emb,
                          float* neigh, float* cnt) {
    int blk = blockIdx.x, d = threadIdx.x;
    int b = blk / Ee, e = blk % Ee;
    int u = eidx[(size_t)b * 2 * Ee + e];
    int v = eidx[(size_t)b * 2 * Ee + Ee + e];
    atomicAdd(&neigh[((size_t)(b * Nn + u)) * DD + d], emb[((size_t)(b * Nn + v)) * DD + d]);
    atomicAdd(&neigh[((size_t)(b * Nn + v)) * DD + d], emb[((size_t)(b * Nn + u)) * DD + d]);
    if (d == 0) { atomicAdd(&cnt[b * Nn + u], 1.f); atomicAdd(&cnt[b * Nn + v], 1.f); }
}

// ---------------- 3. normalize neigh, cast to f16 ----------------
__global__ void k_norm(float* neigh, const float* __restrict__ cnt, _Float16* neigh_h) {
    int i = blockIdx.x, d = threadIdx.x;
    float c = fmaxf(cnt[i], 1.f);
    float v = neigh[(size_t)i * DD + d] / c;
    neigh[(size_t)i * DD + d]   = v;
    neigh_h[(size_t)i * DD + d] = (_Float16)v;
}

// ---------------- 4. transpose+convert 7 128x128 weights to f16 ----------------
__global__ void k_wt(const float* w0, const float* w1, const float* w2, const float* w3,
                     const float* w4, const float* w5, const float* w6, _Float16* dst) {
    const float* ws[7] = {w0, w1, w2, w3, w4, w5, w6};
    int s = blockIdx.x >> 7;       // which weight
    int n = blockIdx.x & 127;      // output column -> row of Wt
    int k = threadIdx.x;
    dst[((size_t)s << 14) + (size_t)n * DD + k] = (_Float16)ws[s][(size_t)k * DD + n];
}

// ---------------- 5. fused dual GEMM: r*(Ax@Wx+bx) + (1-r)*relu(An@Wa+ba) ----------------
// A strips staged into LDS by TDM (wave 0), all 8 waves read fragments via ds_load.
__global__ __launch_bounds__(256) void k_mix(const _Float16* __restrict__ Ax,
                                             const _Float16* __restrict__ An,
                                             const _Float16* __restrict__ WtX,
                                             const _Float16* __restrict__ WtA,
                                             const float* __restrict__ bX,
                                             const float* __restrict__ bA,
                                             const float* __restrict__ rp,
                                             float* outF, _Float16* outH) {
    __shared__ _Float16 sA[2 * 16 * DD];   // only LDS object -> offset 0; [0]=Ax, [1]=An
    int m0   = blockIdx.x * 16;
    int lane = threadIdx.x & 31;
    int n0   = (threadIdx.x >> 5) * 16;
    // Dummy guarded store: never executes (rp is a real device pointer), but it
    // prevents LLVM from proving sA is never written (TDM writes it invisibly)
    // and folding the LDS fragment loads to undef.
    if ((uintptr_t)rp == 1) sA[threadIdx.x] = (_Float16)0;
    if ((threadIdx.x >> 5) == 0) {
        tdm_load_tile16(Ax + (size_t)m0 * DD, 0u);
        tdm_load_tile16(An + (size_t)m0 * DD, (unsigned)(16 * DD * sizeof(_Float16)));
        __builtin_amdgcn_s_wait_tensorcnt(0);
    }
    __syncthreads();
    v8f acc1 = {}; v8f acc2 = {};
    for (int k0 = 0; k0 < DD; k0 += 32) {
        v16h a1 = load_a_frag(sA, 0, k0, lane);
        v16h b1 = load_b_frag(WtX, n0, k0, lane);
        acc1 = __builtin_amdgcn_wmma_f32_16x16x32_f16(false, a1, false, b1,
                                                      (short)0, acc1, false, false);
        v16h a2 = load_a_frag(sA + 16 * DD, 0, k0, lane);
        v16h b2 = load_b_frag(WtA, n0, k0, lane);
        acc2 = __builtin_amdgcn_wmma_f32_16x16x32_f16(false, a2, false, b2,
                                                      (short)0, acc2, false, false);
    }
    float r   = rp[0];
    int  col  = n0 + (lane & 15);
    int  rowb = m0 + (lane >> 4) * 8;
    float bx = bX[col], ba = bA[col];
    for (int j = 0; j < 8; j++) {
        float v1 = acc1[j] + bx;
        float v2 = fmaxf(acc2[j] + ba, 0.f);
        float o  = r * v1 + (1.f - r) * v2;
        size_t idx = (size_t)(rowb + j) * DD + col;
        if (outF) outF[idx] = o;
        if (outH) outH[idx] = (_Float16)o;
    }
}

// ---------------- 6. fe = edge_emb + node_emb[u] + node_emb[v] (f16), zero u_att ----------------
__global__ void k_fe(const float* __restrict__ ef, const int* __restrict__ eidx,
                     const float* __restrict__ We, const float* __restrict__ be,
                     const float* __restrict__ node_emb, _Float16* fe_h, float* u_att) {
    int blk = blockIdx.x, d = threadIdx.x;
    int b = blk / Ee, e = blk % Ee;
    int u = eidx[(size_t)b * 2 * Ee + e];
    int v = eidx[(size_t)b * 2 * Ee + Ee + e];
    const float* f4 = ef + (size_t)blk * 4;
    float val = be[d] + f4[0] * We[d] + f4[1] * We[DD + d] + f4[2] * We[2 * DD + d] +
                f4[3] * We[3 * DD + d] +
                node_emb[((size_t)(b * Nn + u)) * DD + d] +
                node_emb[((size_t)(b * Nn + v)) * DD + d];
    fe_h[(size_t)blk * DD + d] = (_Float16)val;
    if (d == 0) u_att[blk] = 0.f;
}

// ---------------- 7. LSTM step (8 rows, f32 scalar) + q = h@Wq+bq ----------------
__device__ inline float dot128(const float* sh, const float* __restrict__ W, int d) {
    float s = 0.f;
    #pragma unroll 8
    for (int k = 0; k < DD; k++) s += sh[k] * W[(size_t)k * DD + d];
    return s;
}
__device__ inline float sigmoidf(float x) { return 1.f / (1.f + __expf(-x)); }

__global__ void k_lstm(const float* __restrict__ ef, const int* __restrict__ eidx,
                       const float* __restrict__ We, const float* __restrict__ be,
                       const float* __restrict__ node_emb, const int* __restrict__ last_idx,
                       const float* __restrict__ h_in, const float* __restrict__ c_in,
                       const float* Wxi, const float* bxi, const float* Whi, const float* bhi,
                       const float* wci, const float* bci,
                       const float* Wxf, const float* bxf, const float* Whf, const float* bhf,
                       const float* wcf, const float* bcf,
                       const float* Wxc, const float* bxc, const float* Whc, const float* bhc,
                       const float* Wxo, const float* bxo, const float* Who, const float* bho,
                       const float* wco, const float* bco,
                       const float* Wq, const float* bq,
                       float* h_out, float* c_out, float* q_out) {
    __shared__ float xs[DD], hs[DD], cs[DD], cns[DD], hns[DD];
    int b = blockIdx.x, d = threadIdx.x;
    int e = last_idx[b];
    int u = eidx[(size_t)b * 2 * Ee + e];
    int v = eidx[(size_t)b * 2 * Ee + Ee + e];
    const float* f4 = ef + ((size_t)b * Ee + e) * 4;
    xs[d] = be[d] + f4[0] * We[d] + f4[1] * We[DD + d] + f4[2] * We[2 * DD + d] +
            f4[3] * We[3 * DD + d] +
            node_emb[((size_t)(b * Nn + u)) * DD + d] +
            node_emb[((size_t)(b * Nn + v)) * DD + d];
    hs[d] = h_in[b * DD + d];
    cs[d] = c_in[b * DD + d];
    __syncthreads();
    float ig = sigmoidf(dot128(xs, Wxi, d) + bxi[d] + dot128(hs, Whi, d) + bhi[d] +
                        dot128(cs, wci, d) + bci[d]);
    float fg = sigmoidf(dot128(xs, Wxf, d) + bxf[d] + dot128(hs, Whf, d) + bhf[d] +
                        dot128(cs, wcf, d) + bcf[d]);
    float gg = tanhf(dot128(xs, Wxc, d) + bxc[d] + dot128(hs, Whc, d) + bhc[d]);
    float cn = fg * cs[d] + ig * gg;
    cns[d] = cn;
    __syncthreads();
    float og = sigmoidf(dot128(xs, Wxo, d) + bxo[d] + dot128(hs, Who, d) + bho[d] +
                        dot128(cns, wco, d) + bco[d]);
    float hn = og * tanhf(cn);
    hns[d] = hn;
    h_out[b * DD + d] = hn;
    c_out[b * DD + d] = cn;
    __syncthreads();
    q_out[b * DD + d] = dot128(hns, Wq, d) + bq[d];
}

// ---------------- 8. ref = fe@Wref+b fused with u = sum_d tanh(q+ref)*v ----------------
__global__ __launch_bounds__(256) void k_ref_att(const _Float16* __restrict__ feh,
                                                 const _Float16* __restrict__ WtRef,
                                                 const float* __restrict__ bref,
                                                 const float* __restrict__ q,
                                                 const float* __restrict__ vvec,
                                                 float* u_att) {
    __shared__ _Float16 sA[16 * DD];   // only LDS object -> offset 0
    int m0   = blockIdx.x * 16;
    int lane = threadIdx.x & 31;
    int n0   = (threadIdx.x >> 5) * 16;
    // Dummy guarded store (never executes) so the TDM-written LDS reads are not
    // folded to undef; see k_mix.
    if ((uintptr_t)bref == 1) sA[threadIdx.x] = (_Float16)0;
    if ((threadIdx.x >> 5) == 0) {
        tdm_load_tile16(feh + (size_t)m0 * DD, 0u);
        __builtin_amdgcn_s_wait_tensorcnt(0);
    }
    __syncthreads();
    v8f acc = {};
    for (int k0 = 0; k0 < DD; k0 += 32) {
        v16h a  = load_a_frag(sA, 0, k0, lane);
        v16h bf = load_b_frag(WtRef, n0, k0, lane);
        acc = __builtin_amdgcn_wmma_f32_16x16x32_f16(false, a, false, bf,
                                                     (short)0, acc, false, false);
    }
    int col  = n0 + (lane & 15);
    int rowb = m0 + (lane >> 4) * 8;         // rows of a tile never cross a batch (E%16==0)
    float br = bref[col], vv = vvec[col];
    float qc = q[(rowb / Ee) * DD + col];
    for (int j = 0; j < 8; j++) {
        float t = tanhf(qc + acc[j] + br) * vv;
        t += __shfl_xor(t, 1, 16);
        t += __shfl_xor(t, 2, 16);
        t += __shfl_xor(t, 4, 16);
        t += __shfl_xor(t, 8, 16);
        if ((lane & 15) == 0) atomicAdd(&u_att[rowb + j], t);
    }
}

// ---------------- 9. legality mask + softmax over E per batch ----------------
__global__ __launch_bounds__(512) void k_softmax(const float* __restrict__ u_att,
                                                 const int* __restrict__ matched,
                                                 const int* __restrict__ eidx,
                                                 const float* __restrict__ att_scale,
                                                 float* logits, float* probs) {
    __shared__ float red[512];
    __shared__ float s_max, s_sum;
    __shared__ int   s_first;
    int b = blockIdx.x, tid = threadIdx.x;
    int ms = 0;
    for (int n = tid; n < Nn; n += 512) ms += matched[b * Nn + n];
    red[tid] = (float)ms; __syncthreads();
    for (int s = 256; s > 0; s >>= 1) { if (tid < s) red[tid] += red[tid + s]; __syncthreads(); }
    if (tid == 0) s_first = (red[0] == 0.f);
    __syncthreads();
    int isf = s_first;
    float sc = att_scale[0];
    float lmax = -INFINITY;
    for (int e = tid; e < Ee; e += 512) {
        int u = eidx[(size_t)b * 2 * Ee + e];
        int v = eidx[(size_t)b * 2 * Ee + Ee + e];
        bool legal = (matched[b * Nn + u] <= 0) && (matched[b * Nn + v] <= 0);
        float mask = (isf || legal) ? 0.f : -INFINITY;
        float lg = sc * tanhf(u_att[(size_t)b * Ee + e]) + mask;
        logits[(size_t)b * Ee + e] = lg;
        lmax = fmaxf(lmax, lg);
    }
    red[tid] = lmax; __syncthreads();
    for (int s = 256; s > 0; s >>= 1) { if (tid < s) red[tid] = fmaxf(red[tid], red[tid + s]); __syncthreads(); }
    if (tid == 0) s_max = red[0]; __syncthreads();
    float mx = s_max;
    float lsum = 0.f;
    for (int e = tid; e < Ee; e += 512) lsum += __expf(logits[(size_t)b * Ee + e] - mx);
    red[tid] = lsum; __syncthreads();
    for (int s = 256; s > 0; s >>= 1) { if (tid < s) red[tid] += red[tid + s]; __syncthreads(); }
    if (tid == 0) s_sum = red[0]; __syncthreads();
    float inv = 1.f / s_sum;
    for (int e = tid; e < Ee; e += 512)
        probs[(size_t)b * Ee + e] = __expf(logits[(size_t)b * Ee + e] - mx) * inv;
}

// ---------------- launch ----------------
extern "C" void kernel_launch(void* const* d_in, const int* in_sizes, int n_in,
                              void* d_out, int out_size, void* d_ws, size_t ws_size,
                              hipStream_t stream) {
    // ---- inputs (setup_inputs insertion order; each _mk param is (w,b)) ----
    const float* node_coords = (const float*)d_in[0];
    const float* edge_feat   = (const float*)d_in[1];
    const int*   edge_idx    = (const int*)d_in[2];
    const float* h_in        = (const float*)d_in[3];
    const float* c_in        = (const float*)d_in[4];
    const int*   matched     = (const int*)d_in[5];
    const int*   last_idx    = (const int*)d_in[6];
    // params: Wxi(7,8) Whi(9,10) wci(11,12) Wxf(13,14) Whf(15,16) wcf(17,18)
    //         Wxc(19,20) Whc(21,22) Wxo(23,24) Who(25,26) wco(27,28)
    //         W1(29,30) W2(31,32) W3(33,34) agg1(35,36) agg2(37,38) agg3(39,40)
    //         Wq(41,42) Wref(43,44) emb_nodes(45,46) emb_edges(47,48)
    //         v(49) r1(50) r2(51) r3(52) att_scale(53)
    auto W = [&](int i) { return (const float*)d_in[i]; };

    // ---- output layout: probs(B*E) | h(B*D) | c(B*D) | node_emb(B*N*D) ----
    float* out       = (float*)d_out;
    float* probs     = out;
    float* h_out     = out + (size_t)Bb * Ee;
    float* c_out     = h_out + (size_t)Bb * DD;
    float* node_emb  = c_out + (size_t)Bb * DD;

    // ---- workspace layout ----
    const size_t BND = (size_t)Bb * Nn * DD;   // 5,120,000
    const size_t BED = (size_t)Bb * Ee * DD;   // 40,960,000
    float* emb    = (float*)d_ws;
    float* neigh  = emb + BND;
    float* cnt    = neigh + BND;                       // B*N
    float* u_att  = cnt + (size_t)Bb * Nn;             // B*E
    float* logits = u_att + (size_t)Bb * Ee;           // B*E
    float* qbuf   = logits + (size_t)Bb * Ee;          // B*D
    _Float16* emb_h   = (_Float16*)(qbuf + (size_t)Bb * DD);
    _Float16* neigh_h = emb_h + BND;
    _Float16* ctxA_h  = neigh_h + BND;
    _Float16* ctxB_h  = ctxA_h + BND;
    _Float16* fe_h    = ctxB_h + BND;
    _Float16* wt      = fe_h + BED;                    // 7 x 128x128 transposed f16
    (void)ws_size; (void)n_in; (void)in_sizes; (void)out_size;

    // 1. node embedding + zero neigh/cnt
    k_embed<<<Bb * Nn, DD, 0, stream>>>(node_coords, W(45), W(46), emb, emb_h, neigh, cnt);
    // 2. neighbor scatter-add
    k_scatter<<<Bb * Ee, DD, 0, stream>>>(edge_idx, emb, neigh, cnt);
    // 3. normalize
    k_norm<<<Bb * Nn, DD, 0, stream>>>(neigh, cnt, neigh_h);
    // 4. f16 transposed weights: W1 W2 W3 agg1 agg2 agg3 Wref
    k_wt<<<7 * 128, DD, 0, stream>>>(W(29), W(31), W(33), W(35), W(37), W(39), W(43), wt);
    // 5-7. three fused dual-GEMM GNN layers (WMMA + TDM-staged A strips)
    const int gridN = (Bb * Nn) / 16;  // 2500
    k_mix<<<gridN, 256, 0, stream>>>(emb_h,  neigh_h, wt + 0 * 16384, wt + 3 * 16384,
                                     W(30), W(36), W(50), nullptr, ctxA_h);
    k_mix<<<gridN, 256, 0, stream>>>(ctxA_h, neigh_h, wt + 1 * 16384, wt + 4 * 16384,
                                     W(32), W(38), W(51), nullptr, ctxB_h);
    k_mix<<<gridN, 256, 0, stream>>>(ctxB_h, neigh_h, wt + 2 * 16384, wt + 5 * 16384,
                                     W(34), W(40), W(52), node_emb, nullptr);
    // 8. fe (f16) + zero u_att
    k_fe<<<Bb * Ee, DD, 0, stream>>>(edge_feat, edge_idx, W(47), W(48), node_emb, fe_h, u_att);
    // 9. LSTM + q
    k_lstm<<<Bb, DD, 0, stream>>>(edge_feat, edge_idx, W(47), W(48), node_emb, last_idx,
                                  h_in, c_in,
                                  W(7), W(8), W(9), W(10), W(11), W(12),
                                  W(13), W(14), W(15), W(16), W(17), W(18),
                                  W(19), W(20), W(21), W(22),
                                  W(23), W(24), W(25), W(26), W(27), W(28),
                                  W(41), W(42),
                                  h_out, c_out, qbuf);
    // 10. Wref GEMM fused with attention reduction (WMMA + TDM-staged fe strip)
    k_ref_att<<<(Bb * Ee) / 16, 256, 0, stream>>>(fe_h, wt + 6 * 16384, W(44), qbuf,
                                                  W(49), u_att);
    // 11. masked softmax
    k_softmax<<<Bb, 512, 0, stream>>>(u_att, matched, edge_idx, W(53), logits, probs);
}
